// ScaledDotProductAttention_32255204393471
// MI455X (gfx1250) — matrix operand order repaired and checked by reading, not compiled, not run on hardware
//
#include <hip/hip_runtime.h>

typedef __attribute__((ext_vector_type(16))) _Float16 v16h;
typedef __attribute__((ext_vector_type(8)))  _Float16 h8;
typedef __attribute__((ext_vector_type(4)))  _Float16 h4;
typedef __attribute__((ext_vector_type(8)))  float    v8f;

#define Bx 2
#define Hx 16
#define Sx 2048
#define Dx 128
#define BM 128                 // query rows per block (8 waves x 16 rows)
#define BN 64                  // keys per inner iteration (4 x 16-key tiles)
#define NKB (Sx / BN)          // 32 iterations
#define KPAD 136               // Dx + 8 halves: 272B rows, 16B aligned, 68 dwords
#define VPAD 72                // BN + 8 halves: 144B rows, 16B aligned, 36 dwords
#define PPAD 72

#define WMMA_F16(A, B, C) \
  __builtin_amdgcn_wmma_f32_16x16x32_f16(false, (A), false, (B), (short)0, (C), false, false)

// 32-byte (16 x f16) fragment chunk as two 16B LDS loads.
__device__ __forceinline__ v16h ld_frag32(const _Float16* p) {
  union { v16h v; h8 h[2]; } u;
  u.h[0] = *(const h8*)(p);
  u.h[1] = *(const h8*)(p + 8);
  return u.v;
}

// xor-butterfly max over the 16-lane half, entirely on the VALU via v_permlane16_b32.
// Lane-select nibbles encode src = lane ^ mask for masks 1,2,4,8.
__device__ __forceinline__ float half16_max(float x) {
  union { float f; int i; } a, b;
  a.f = x; b.i = __builtin_amdgcn_permlane16(a.i, a.i, 0x67452301u, 0xEFCDAB89u, false, false);
  x = fmaxf(x, b.f);
  a.f = x; b.i = __builtin_amdgcn_permlane16(a.i, a.i, 0x54761032u, 0xDCFE98BAu, false, false);
  x = fmaxf(x, b.f);
  a.f = x; b.i = __builtin_amdgcn_permlane16(a.i, a.i, 0x32107654u, 0xBA98FEDCu, false, false);
  x = fmaxf(x, b.f);
  a.f = x; b.i = __builtin_amdgcn_permlane16(a.i, a.i, 0xFEDCBA98u, 0x76543210u, false, false);
  return fmaxf(x, b.f);
}

struct KStage { float4 r[8]; };
struct VStage { float2 r[16]; };

__device__ __forceinline__ void load_k(KStage& s, const float4* kt, int tid) {
  #pragma unroll
  for (int j = 0; j < 8; ++j) s.r[j] = kt[j * 256 + tid];
}
__device__ __forceinline__ void store_k(const KStage& s, _Float16 (*Kb)[KPAD], int tid) {
  #pragma unroll
  for (int j = 0; j < 8; ++j) {
    int idx = j * 256 + tid, row = idx >> 5, c4 = idx & 31;
    h4 hk = { (_Float16)s.r[j].x, (_Float16)s.r[j].y, (_Float16)s.r[j].z, (_Float16)s.r[j].w };
    *(h4*)&Kb[row][c4 * 4] = hk;
  }
}
// V handled as units of (4 consecutive keys) x (2 consecutive d): 16 coalesced b64
// global loads, then 8 packed b64 LDS stores into the transposed Vsh[d][key] tile.
__device__ __forceinline__ void load_v(VStage& s, const float* vbase, int tid) {
  #pragma unroll
  for (int j = 0; j < 4; ++j) {
    int u = j * 256 + tid, dp = u & 63, kq = u >> 6;   // d-pair 0..63, key-quad 0..15
    const float* p = vbase + (size_t)(4 * kq) * Dx + 2 * dp;
    #pragma unroll
    for (int e = 0; e < 4; ++e) s.r[j * 4 + e] = *(const float2*)(p + (size_t)e * Dx);
  }
}
__device__ __forceinline__ void store_v(const VStage& s, _Float16 (*Vb)[VPAD], int tid) {
  #pragma unroll
  for (int j = 0; j < 4; ++j) {
    int u = j * 256 + tid, dp = u & 63, kq = u >> 6;
    h4 d0 = { (_Float16)s.r[j*4+0].x, (_Float16)s.r[j*4+1].x,
              (_Float16)s.r[j*4+2].x, (_Float16)s.r[j*4+3].x };
    h4 d1 = { (_Float16)s.r[j*4+0].y, (_Float16)s.r[j*4+1].y,
              (_Float16)s.r[j*4+2].y, (_Float16)s.r[j*4+3].y };
    *(h4*)&Vb[2 * dp][4 * kq]     = d0;   // Vsh[d][key] = V[key][d]
    *(h4*)&Vb[2 * dp + 1][4 * kq] = d1;
  }
}

__global__ __launch_bounds__(256)
void fa_fwd(const float* __restrict__ q, const float* __restrict__ k,
            const float* __restrict__ v, float* __restrict__ out)
{
  __shared__ _Float16 Ksh[2][BN][KPAD];     // K tiles, row-major [key][d]
  __shared__ _Float16 Vsh[2][Dx][VPAD];     // V tiles transposed [d][key]
  __shared__ _Float16 Psh[8][16][PPAD];     // per-wave P relayout scratch (16 x 64)

  const int tid  = threadIdx.x;
  const int wave = tid >> 5;
  const int lane = tid & 31;
  const int l15  = lane & 15;
  const int hi   = lane >> 4;
  const int koff = hi * 8;
  const float scale = 0.088388347648318447f;   // 1/sqrt(128)

  const int bh    = blockIdx.y;
  const int qrow0 = blockIdx.x * BM + wave * 16;
  const size_t base = (size_t)bh * Sx;

  // ---- Q A-fragments (16x32 f16 per 32-wide d-chunk), scale folded in ----
  v16h qa[4];
  {
    const float* qp = q + (base + qrow0 + l15) * Dx;
    #pragma unroll
    for (int c = 0; c < 4; ++c) {
      float t[16];
      const float4* pA = (const float4*)(qp + 32 * c + 16 * hi);
      const float4* pB = (const float4*)(qp + 32 * c + 16 * hi + 8);
      float4 a0 = pA[0], a1 = pA[1], b0 = pB[0], b1 = pB[1];
      t[0]=a0.x; t[1]=a0.y; t[2]=a0.z;  t[3]=a0.w;
      t[4]=a1.x; t[5]=a1.y; t[6]=a1.z;  t[7]=a1.w;
      t[8]=b0.x; t[9]=b0.y; t[10]=b0.z; t[11]=b0.w;
      t[12]=b1.x; t[13]=b1.y; t[14]=b1.z; t[15]=b1.w;
      #pragma unroll
      for (int i = 0; i < 16; ++i) qa[c][i] = (_Float16)(t[i] * scale);
    }
  }

  v16h ones;                                 // all-ones B: row-sum of P via WMMA
  #pragma unroll
  for (int i = 0; i < 16; ++i) ones[i] = (_Float16)1.0f;

  v8f acc[8] = {};
  float mrow[8], lrow[8];
  #pragma unroll
  for (int r = 0; r < 8; ++r) { mrow[r] = -3.0e38f; lrow[r] = 0.0f; }

  const float4* kp4 = (const float4*)(k + base * Dx);
  const float*  vp  = v + base * Dx;

  // ---- prologue: stage tile 0 into buffer 0 ----
  {
    KStage ks; VStage vs;
    load_k(ks, kp4, tid);  load_v(vs, vp, tid);
    store_k(ks, Ksh[0], tid);  store_v(vs, Vsh[0], tid);
  }
  __syncthreads();

  for (int kb = 0; kb < NKB; ++kb) {
    const int cur = kb & 1;
    const int nxt = cur ^ 1;
    const bool more = (kb + 1 < NKB);        // uniform: EXEC stays all-ones

    // ---- issue K global loads for tile kb+1 ----
    KStage ks;
    if (more) {
      const float4* kt = kp4 + (size_t)(kb + 1) * (BN * Dx / 4);
      load_k(ks, kt, tid);
      if (kb + 2 < NKB)
        __builtin_prefetch((const char*)(kt + BN * Dx / 4) + tid * 128, 0, 1);
    }

    // ---- S = Q K^T : 4 key tiles x 4 d-chunks = 16 WMMAs ----
    v8f s[4] = {};
    #pragma unroll
    for (int c = 0; c < 4; ++c) {
      v16h kf0 = ld_frag32(&Ksh[cur][ 0 + l15][32 * c + 16 * hi]);
      v16h kf1 = ld_frag32(&Ksh[cur][16 + l15][32 * c + 16 * hi]);
      v16h kf2 = ld_frag32(&Ksh[cur][32 + l15][32 * c + 16 * hi]);
      v16h kf3 = ld_frag32(&Ksh[cur][48 + l15][32 * c + 16 * hi]);
      s[0] = WMMA_F16(qa[c], kf0, s[0]);
      s[1] = WMMA_F16(qa[c], kf1, s[1]);
      s[2] = WMMA_F16(qa[c], kf2, s[2]);
      s[3] = WMMA_F16(qa[c], kf3, s[3]);
    }

    // ---- issue V global loads for tile kb+1 (hide under softmax + PV) ----
    VStage vs;
    if (more) {
      const float* vt = vp + (size_t)(kb + 1) * (BN * Dx);
      load_v(vs, vt, tid);
      if (kb + 2 < NKB)
        __builtin_prefetch((const char*)(vt + BN * Dx) + tid * 128, 0, 1);
    }

    // ---- online softmax: max on VALU (permlane16), sum via WMMA below ----
    float alpha[8];
    #pragma unroll
    for (int r = 0; r < 8; ++r) {
      float mx = fmaxf(fmaxf(s[0][r], s[1][r]), fmaxf(s[2][r], s[3][r]));
      mx = half16_max(mx);
      float mnew = fmaxf(mrow[r], mx);
      float a = __expf(mrow[r] - mnew);
      alpha[r] = a;
      mrow[r] = mnew;
      int prow = r + 8 * hi;
      Psh[wave][prow][l15]      = (_Float16)__expf(s[0][r] - mnew);
      Psh[wave][prow][l15 + 16] = (_Float16)__expf(s[1][r] - mnew);
      Psh[wave][prow][l15 + 32] = (_Float16)__expf(s[2][r] - mnew);
      Psh[wave][prow][l15 + 48] = (_Float16)__expf(s[3][r] - mnew);
      #pragma unroll
      for (int n = 0; n < 8; ++n) acc[n][r] *= a;
    }

    // ---- re-read P as two 16x32 A-fragments ----
    v16h pa0, pa1;
    {
      const _Float16* pp = &Psh[wave][l15][0];
      union { v16h vv; h8 h[2]; } u;
      u.h[0] = *(const h8*)(pp + 16 * hi);
      u.h[1] = *(const h8*)(pp + 16 * hi + 8);
      pa0 = u.vv;
      u.h[0] = *(const h8*)(pp + 32 + 16 * hi);
      u.h[1] = *(const h8*)(pp + 32 + 16 * hi + 8);
      pa1 = u.vv;
    }

    // ---- row sums of P via WMMA against all-ones B ----
    v8f rs = {};
    rs = WMMA_F16(pa0, ones, rs);
    rs = WMMA_F16(pa1, ones, rs);
    #pragma unroll
    for (int r = 0; r < 8; ++r) lrow[r] = lrow[r] * alpha[r] + rs[r];

    // ---- O += P V : 8 d-tiles x 2 key-chunks = 16 WMMAs ----
    #pragma unroll
    for (int n = 0; n < 8; n += 2) {
      v16h a0 = ld_frag32(&Vsh[cur][16 * n + l15][16 * hi]);
      v16h a1 = ld_frag32(&Vsh[cur][16 * n + l15][32 + 16 * hi]);
      v16h b0 = ld_frag32(&Vsh[cur][16 * (n + 1) + l15][16 * hi]);
      v16h b1 = ld_frag32(&Vsh[cur][16 * (n + 1) + l15][32 + 16 * hi]);
      acc[n]     = WMMA_F16(pa0, a0, acc[n]);
      acc[n]     = WMMA_F16(pa1, a1, acc[n]);
      acc[n + 1] = WMMA_F16(pa0, b0, acc[n + 1]);
      acc[n + 1] = WMMA_F16(pa1, b1, acc[n + 1]);
    }

    // ---- convert + store tile kb+1 into the free buffer ----
    if (more) {
      store_k(ks, Ksh[nxt], tid);
      store_v(vs, Vsh[nxt], tid);
    }
    __syncthreads();
  }

  // ---- epilogue: normalize by row sums and store fp32 ----
  #pragma unroll
  for (int r = 0; r < 8; ++r) {
    float inv = 1.0f / lrow[r];
    float* op = out + (base + qrow0 + r + 8 * hi) * Dx;
    #pragma unroll
    for (int n = 0; n < 8; ++n) op[16 * n + l15] = acc[n][r] * inv;
  }
}

extern "C" void kernel_launch(void* const* d_in, const int* in_sizes, int n_in,
                              void* d_out, int out_size, void* d_ws, size_t ws_size,
                              hipStream_t stream) {
  const float* q = (const float*)d_in[0];
  const float* k = (const float*)d_in[1];
  const float* v = (const float*)d_in[2];
  float* out = (float*)d_out;
  dim3 grid(Sx / BM, Bx * Hx);   // (16, 32): one 128-row tile per (b,h) per block
  fa_fwd<<<grid, dim3(256), 0, stream>>>(q, k, v, out);
}
